// CrossAttention_86887188398083
// MI455X (gfx1250) — matrix operand order repaired and checked by reading, not compiled, not run on hardware
//
#include <hip/hip_runtime.h>

typedef __bf16 bf16;
typedef bf16  v8bf  __attribute__((ext_vector_type(8)));
typedef bf16  v16bf __attribute__((ext_vector_type(16)));
typedef float v8f   __attribute__((ext_vector_type(8)));
typedef unsigned int v4u __attribute__((ext_vector_type(4)));
typedef int   v8i   __attribute__((ext_vector_type(8)));
typedef int   v4i   __attribute__((ext_vector_type(4)));

#define WMMA_BF16(a, b, c) \
  __builtin_amdgcn_wmma_f32_16x16x32_bf16(false, (a), false, (b), (short)0, (c), false, false)

constexpr int Dk  = 512;
constexpr int LQ  = 4096;
constexpr int LKV = 4096;
constexpr int NB  = 4;

__device__ __forceinline__ v16bf ldfrag(const bf16* lo, const bf16* hi) {
  v8bf a = *(const v8bf*)lo;
  v8bf b = *(const v8bf*)hi;
  v16bf r;
  for (int i = 0; i < 8; i++) { r[i] = a[i]; r[8 + i] = b[i]; }
  return r;
}

// ---------------------------------------------------------------------------
// TDM: DMA one 32-row x 512-col bf16 tile (memory row stride = Dk elems)
// into LDS, padding +16B per 1024B row so LDS row stride = 1040B
// (260 dwords -> conflict-free ds_load_b128 B-fragments).
// ---------------------------------------------------------------------------
__device__ __forceinline__ void tdm_load_k_tile(const bf16* gsrc, unsigned lds_off) {
  unsigned long long ga = (unsigned long long)(__SIZE_TYPE__)gsrc;
  v4u g0;
  g0[0] = 1u;                                   // count=1 (valid user descriptor)
  g0[1] = lds_off;                              // lds_addr (bytes)
  g0[2] = (unsigned)(ga & 0xFFFFFFFFu);         // global_addr[31:0]
  g0[3] = ((unsigned)(ga >> 32) & 0x01FFFFFFu)  // global_addr[56:32]
          | (2u << 30);                         // type=2 ("image")
  v8i g1;
  g1[0] = (1 << 16)                             // data_size=1 -> 2 bytes
        | (1 << 20)                             // pad_enable
        | (7 << 22)                             // pad_interval: 256 dwords (1024B)
        | (3 << 25);                            // pad_amount:   4 dwords (16B)
  g1[1] = (int)((unsigned)(Dk & 0xFFFF) << 16);                 // tensor_dim0 lo16
  g1[2] = (int)(((unsigned)Dk >> 16) | ((unsigned)(LKV & 0xFFFF) << 16)); // dim0 hi | dim1 lo
  g1[3] = (int)(((unsigned)LKV >> 16) | ((unsigned)Dk << 16));  // dim1 hi | tile_dim0=512
  g1[4] = 32;                                   // tile_dim1=32, tile_dim2=0
  g1[5] = Dk;                                   // tensor_dim0_stride lo32 (=512 elems)
  g1[6] = 0;
  g1[7] = 0;
  v4i z4 = {0, 0, 0, 0};
  v8i z8 = {0, 0, 0, 0, 0, 0, 0, 0};
  __builtin_amdgcn_tensor_load_to_lds(g0, g1, z4, z4, z8, 0);
}

// ---------------------------------------------------------------------------
// f32 -> bf16 bulk convert (weights)
// ---------------------------------------------------------------------------
__global__ __launch_bounds__(256) void cvt_f32_bf16(const float* __restrict__ src,
                                                    bf16* __restrict__ dst, int n) {
  int i = (blockIdx.x * 256 + threadIdx.x) * 4;
  if (i + 3 < n) {
    dst[i + 0] = (bf16)src[i + 0];
    dst[i + 1] = (bf16)src[i + 1];
    dst[i + 2] = (bf16)src[i + 2];
    dst[i + 3] = (bf16)src[i + 3];
  }
}

// ---------------------------------------------------------------------------
// Generic 16x16-tile WMMA GEMM:  Out = (A @ W^T + bias) * scale
//   A: [M,K] f32 or bf16; W: [N,K] bf16 (pre-converted); bias: [N] f32
//   TRANS_OUT: Out as [batch][N][Lrows] bf16 (V^T); OUT_F32: f32 row-major.
// ---------------------------------------------------------------------------
template <bool A_BF16, bool TRANS_OUT, bool OUT_F32>
__global__ __launch_bounds__(128) void gemm_wmma(
    const void* __restrict__ Av, const bf16* __restrict__ W,
    const float* __restrict__ bias, void* __restrict__ Outv,
    float scale, int M, int N, int K, int Lrows) {
  const int lane = threadIdx.x & 31;
  const int wave = threadIdx.x >> 5;
  const int half = lane >> 4;
  const int l15  = lane & 15;

  const int m0 = blockIdx.x * 16;
  const int n0 = blockIdx.y * 64 + wave * 16;

  const int arow = m0 + l15;
  const int wrow = n0 + l15;

  v8f acc = {};
  for (int k0 = 0; k0 < K; k0 += 32) {
    v16bf a;
    const int akb = k0 + half * 8;
    if (A_BF16) {
      const bf16* A = (const bf16*)Av;
      a = ldfrag(A + (size_t)arow * K + akb, A + (size_t)arow * K + akb + 16);
    } else {
      const float* A  = (const float*)Av;
      const float* p0 = A + (size_t)arow * K + akb;
      for (int i = 0; i < 8; i++) { a[i] = (bf16)p0[i]; a[8 + i] = (bf16)p0[16 + i]; }
    }
    const bf16* wp = W + (size_t)wrow * K + k0 + half * 8;
    v16bf b = ldfrag(wp, wp + 16);
    acc = WMMA_BF16(a, b, acc);
  }

  const float bv = bias[n0 + l15];

  if (TRANS_OUT) {
    bf16* Out  = (bf16*)Outv;
    int   bidx = m0 / Lrows;
    int   mloc = m0 - bidx * Lrows;
    v8bf  pk;
    for (int r = 0; r < 8; r++) pk[r] = (bf16)((acc[r] + bv) * scale);
    *(v8bf*)(Out + (size_t)bidx * N * Lrows + (size_t)(n0 + l15) * Lrows +
             mloc + half * 8) = pk;
  } else if (OUT_F32) {
    float* Out = (float*)Outv;
    for (int r = 0; r < 8; r++)
      Out[(size_t)(m0 + half * 8 + r) * N + n0 + l15] = (acc[r] + bv) * scale;
  } else {
    bf16* Out = (bf16*)Outv;
    for (int r = 0; r < 8; r++)
      Out[(size_t)(m0 + half * 8 + r) * N + n0 + l15] = (bf16)((acc[r] + bv) * scale);
  }
}

// ---------------------------------------------------------------------------
// Flash attention.  Block = 4 waves <-> one (batch, 16-row q tile).
//  * K tiles (32x512 bf16) DMA'd to LDS by the Tensor Data Mover, double-
//    buffered, TENSORcnt-synchronized, shared by all 4 waves.
//  * S = Q K^T split over d: each wave computes a 128-wide partial (8 WMMAs),
//    partials reduced through LDS -> no redundant matrix work.
//  * Online softmax per wave (identical, VALU-only); P re-laid-out C->A via
//    a small LDS bounce; each wave accumulates its 128-column ctx slice
//    against transposed V (contiguous global fragments).
// ---------------------------------------------------------------------------
__global__ __launch_bounds__(128) void attn_wmma(
    const bf16* __restrict__ Qb, const bf16* __restrict__ Kb,
    const bf16* __restrict__ Vt, bf16* __restrict__ Ctx) {
  constexpr int QSTR = 520;   // Q LDS row stride (bf16 elems), padded
  constexpr int KSTR = 520;   // K LDS row stride: 512 + TDM pad (4 dwords)
  constexpr int PSTR = 40;

  __shared__ bf16  Qs[16 * QSTR];
  __shared__ bf16  Ks[2][32 * KSTR];
  __shared__ bf16  Ps[16 * PSTR];
  __shared__ float Sred[8][32 * 8];   // [wave*2 + tile][lane*8 + r]

  const int lane = threadIdx.x & 31;
  const int wave = threadIdx.x >> 5;
  const int half = lane >> 4;
  const int l15  = lane & 15;

  const int b  = blockIdx.x / (LQ / 16);
  const int q0 = (blockIdx.x % (LQ / 16)) * 16;

  const bf16* Qg = Qb + ((size_t)b * LQ + q0) * Dk;
  const bf16* Kg = Kb + (size_t)b * LKV * Dk;
  const bf16* Vg = Vt + (size_t)b * Dk * LKV;

  // Stage the 16x512 Q tile into LDS.
  for (int i = threadIdx.x; i < 16 * 64; i += 128) {
    int r = i >> 6, c = (i & 63) * 8;
    *(v8bf*)(Qs + r * QSTR + c) = *(const v8bf*)(Qg + (size_t)r * Dk + c);
  }
  // Prologue: TDM for the first K tile.
  if (wave == 0)
    tdm_load_k_tile(Kg, (unsigned)(__SIZE_TYPE__)&Ks[0][0]);
  __syncthreads();

  const int d0 = wave * 128;  // this wave's d-slice (S partial + ctx columns)

  float m[8], l[8];
  v8f   o[8];
  for (int r = 0; r < 8; r++) { m[r] = -1e30f; l[r] = 0.f; }
  for (int t = 0; t < 8; t++) { v8f z = {}; o[t] = z; }

  constexpr int NSTEP = LKV / 32;
  for (int step = 0; step < NSTEP; step++) {
    const int kv0 = step * 32;
    const int cur = step & 1;

    __syncthreads();  // B1: K[cur^1]/Sred/Ps no longer being read
    if (wave == 0) {
      if (step + 1 < NSTEP)
        tdm_load_k_tile(Kg + (size_t)(kv0 + 32) * Dk,
                        (unsigned)(__SIZE_TYPE__)&Ks[cur ^ 1][0]);
      if (step + 1 < NSTEP) __builtin_amdgcn_s_wait_tensorcnt(1);
      else                  __builtin_amdgcn_s_wait_tensorcnt(0);
    }
    __syncthreads();  // B2: K[cur] resident in LDS

    // ---- partial S over this wave's 128 d-columns: 2 tiles x 4 chunks
    v8f s0 = {}, s1 = {};
    const bf16* krow0 = &Ks[cur][0] + (size_t)l15 * KSTR + d0 + half * 16;
    const bf16* krow1 = krow0 + (size_t)16 * KSTR;
    for (int dc = 0; dc < 4; dc++) {
      const bf16* qp = Qs + l15 * QSTR + d0 + dc * 32 + half * 16;
      v16bf a  = ldfrag(qp, qp + 8);
      v16bf b0 = ldfrag(krow0 + dc * 32, krow0 + dc * 32 + 8);
      v16bf b1 = ldfrag(krow1 + dc * 32, krow1 + dc * 32 + 8);
      s0 = WMMA_BF16(a, b0, s0);
      s1 = WMMA_BF16(a, b1, s1);
    }
    *(v8f*)&Sred[wave * 2 + 0][lane * 8] = s0;
    *(v8f*)&Sred[wave * 2 + 1][lane * 8] = s1;
    __syncthreads();  // B3: all partials written

    v8f f0 = {}, f1 = {};
    for (int w = 0; w < 4; w++) {
      f0 += *(const v8f*)&Sred[w * 2 + 0][lane * 8];
      f1 += *(const v8f*)&Sred[w * 2 + 1][lane * 8];
    }

    // ---- online softmax (identical in every wave; rows = VGPR index)
    for (int r = 0; r < 8; r++) {
      float mx = fmaxf(f0[r], f1[r]);
      for (int off = 8; off >= 1; off >>= 1) mx = fmaxf(mx, __shfl_xor(mx, off, 32));
      float mn    = fmaxf(m[r], mx);
      float alpha = __expf(m[r] - mn);
      float p0    = __expf(f0[r] - mn);
      float p1    = __expf(f1[r] - mn);
      float rs    = p0 + p1;
      for (int off = 8; off >= 1; off >>= 1) rs += __shfl_xor(rs, off, 32);
      l[r] = l[r] * alpha + rs;
      m[r] = mn;
      for (int t = 0; t < 8; t++) o[t][r] *= alpha;
      if (wave == 0) {
        bf16* pr     = Ps + (r + 8 * half) * PSTR;
        pr[l15]      = (bf16)p0;
        pr[16 + l15] = (bf16)p1;
      }
    }
    __syncthreads();  // B4: P visible

    // ---- P A-fragment from LDS bounce
    const bf16* pp = Ps + l15 * PSTR + half * 16;
    v16bf pa = ldfrag(pp, pp + 8);

    // ---- ctx += P @ V for this wave's 128 columns (V^T: contiguous frags)
    const bf16* vbase = Vg + (size_t)(d0 + l15) * LKV + kv0 + half * 16;
    __builtin_prefetch(vbase + 32, 0, 3);
    for (int t = 0; t < 8; t++) {
      v16bf bv = *(const v16bf*)(vbase + (size_t)t * 16 * LKV);
      o[t] = WMMA_BF16(pa, bv, o[t]);
    }
  }

  // ---- normalize and store ctx slice (bf16, feeds the final projection)
  bf16* cg = Ctx + ((size_t)b * LQ + q0) * Dk;
  for (int r = 0; r < 8; r++) {
    float inv = 1.f / l[r];
    for (int t = 0; t < 8; t++)
      cg[(size_t)(8 * half + r) * Dk + d0 + t * 16 + l15] = (bf16)(o[t][r] * inv);
  }
}

// ---------------------------------------------------------------------------
extern "C" void kernel_launch(void* const* d_in, const int* in_sizes, int n_in,
                              void* d_out, int out_size, void* d_ws, size_t ws_size,
                              hipStream_t stream) {
  const float* x  = (const float*)d_in[0];
  const float* y  = (const float*)d_in[1];
  const float* Wq = (const float*)d_in[2];
  const float* bq = (const float*)d_in[3];
  const float* Wk = (const float*)d_in[4];
  const float* bk = (const float*)d_in[5];
  const float* Wv = (const float*)d_in[6];
  const float* bv = (const float*)d_in[7];
  const float* Wo = (const float*)d_in[8];
  const float* bo = (const float*)d_in[9];
  float* out = (float*)d_out;

  char*  ws = (char*)d_ws;
  size_t sz = (size_t)NB * 4096 * 512 * sizeof(bf16);  // 16 MB
  size_t wz = (size_t)Dk * Dk * sizeof(bf16);          // 512 KB
  bf16* Qbuf = (bf16*)(ws);
  bf16* Kbuf = (bf16*)(ws + sz);
  bf16* Vtbf = (bf16*)(ws + 2 * sz);
  bf16* Cbuf = (bf16*)(ws + 3 * sz);
  bf16* Wqb  = (bf16*)(ws + 4 * sz);
  bf16* Wkb  = (bf16*)(ws + 4 * sz + wz);
  bf16* Wvb  = (bf16*)(ws + 4 * sz + 2 * wz);
  bf16* Wob  = (bf16*)(ws + 4 * sz + 3 * wz);

  const int nW = Dk * Dk;  // 262144
  cvt_f32_bf16<<<dim3(nW / 1024), dim3(256), 0, stream>>>(Wq, Wqb, nW);
  cvt_f32_bf16<<<dim3(nW / 1024), dim3(256), 0, stream>>>(Wk, Wkb, nW);
  cvt_f32_bf16<<<dim3(nW / 1024), dim3(256), 0, stream>>>(Wv, Wvb, nW);
  cvt_f32_bf16<<<dim3(nW / 1024), dim3(256), 0, stream>>>(Wo, Wob, nW);

  const int   M      = NB * LQ;  // 16384
  const float qscale = 0.04419417382415922f;  // 1/sqrt(512)
  dim3 blk(128);
  dim3 grid(M / 16, Dk / 64);

  gemm_wmma<false, false, false><<<grid, blk, 0, stream>>>(x, Wqb, bq, Qbuf, qscale, M, Dk, Dk, LQ);
  gemm_wmma<false, false, false><<<grid, blk, 0, stream>>>(y, Wkb, bk, Kbuf, 1.f, M, Dk, Dk, LKV);
  gemm_wmma<false, true,  false><<<grid, blk, 0, stream>>>(y, Wvb, bv, Vtbf, 1.f, M, Dk, Dk, LKV);

  attn_wmma<<<dim3(NB * LQ / 16), blk, 0, stream>>>(Qbuf, Kbuf, Vtbf, Cbuf);

  gemm_wmma<true, false, true><<<grid, blk, 0, stream>>>(Cbuf, Wob, bo, out, 1.f, M, Dk, Dk, LQ);
}
